// SelfAttention_36687610643187
// MI455X (gfx1250) — matrix operand order repaired and checked by reading, not compiled, not run on hardware
//
#include <hip/hip_runtime.h>
#include <hip/hip_bf16.h>

typedef __bf16 bf16;
typedef __attribute__((ext_vector_type(16))) __bf16 v16bf;
typedef __attribute__((ext_vector_type(8)))  __bf16 v8bf;
typedef __attribute__((ext_vector_type(8)))  float  v8f;

#define B_  4
#define S_  2048
#define C_  1024
#define H_  16
#define CH_ 64

// ---- CDNA5 async global->LDS copy (ASYNCcnt path, CDNA5 ISA)
__device__ inline void async_cp16(unsigned lds_off, const bf16* g) {
    asm volatile("global_load_async_to_lds_b128 %0, %1, off"
                 :: "v"(lds_off), "v"((unsigned long long)(const void*)g)
                 : "memory");
}
#define WAIT_ASYNC(imm) asm volatile("s_wait_asynccnt " #imm ::: "memory")

// generic LDS pointer -> wave-relative LDS byte address (aperture keeps addr[31:0])
__device__ inline unsigned lds_addr(const void* p) {
    return (unsigned)(unsigned long long)p;
}

__device__ inline v8f wmma_bf16(v16bf a, v16bf b, v8f c) {
    return __builtin_amdgcn_wmma_f32_16x16x32_bf16(
        false, a, false, b, (short)0, c, false, false);
}

// A-fragment: 16x32 bf16, row-major source, stride ld elements.
__device__ inline v16bf load_a_frag(const bf16* base, size_t ld) {
    const int lane = threadIdx.x & 31;
    const int m = lane & 15, hi = lane >> 4;
    const bf16* p = base + (size_t)m * ld + hi * 8;
    v8bf lo = *(const v8bf*)(p);
    v8bf hq = *(const v8bf*)(p + 16);
    return __builtin_shufflevector(lo, hq, 0,1,2,3,4,5,6,7,8,9,10,11,12,13,14,15);
}

// B-fragment: 32x16 bf16; column n of B = row n of a row-major NxK matrix.
__device__ inline v16bf load_b_frag(const bf16* base, size_t ld) {
    const int lane = threadIdx.x & 31;
    const int n = lane & 15, hi = lane >> 4;
    const bf16* p = base + (size_t)n * ld + hi * 8;
    v8bf lo = *(const v8bf*)(p);
    v8bf hq = *(const v8bf*)(p + 16);
    return __builtin_shufflevector(lo, hq, 0,1,2,3,4,5,6,7,8,9,10,11,12,13,14,15);
}

// ---------------------------------------------------------------- convert
__global__ void __launch_bounds__(256) cvt_f32_bf16(const float* __restrict__ in,
                                                    bf16* __restrict__ out, int n) {
    int i = blockIdx.x * 256 + threadIdx.x;
    if (i < n) out[i] = (bf16)in[i];
}

// ---------------------------------------------------------------- GEMM core
// C[M x N] = A[M x K] * W[N x K]^T, bf16 in, f32 acc.
// 256 threads = 8 waves (4 x 2); block tile 128(M) x 128(N); BK=32.
// Wave tile 32 x 64: 2 A-frags x 4 B-frags = 8 WMMA / k-step.
// Double-buffered async global->LDS staging; last iteration peeled (no branchy waits).
template <typename StoreFn>
__device__ inline void gemm_body(const bf16* __restrict__ A, const bf16* __restrict__ W,
                                 int K, StoreFn&& store) {
    __shared__ __attribute__((aligned(16))) bf16 Abuf[2][128 * 32];
    __shared__ __attribute__((aligned(16))) bf16 Bbuf[2][128 * 32];

    const int tid  = threadIdx.x;
    const int wave = tid >> 5;
    const int wr = wave >> 1, wc = wave & 1;
    const int m0 = blockIdx.y * 128;
    const int n0 = blockIdx.x * 128;

    // copy mapping: each tile 128 rows x 32 cols -> 32B (2 x b128) per thread per tile
    const int crow = tid >> 1, cseg = (tid & 1) * 16;

    auto issue = [&](int buf, int k0) {
        unsigned ao = lds_addr(&Abuf[buf][crow * 32 + cseg]);
        const bf16* as = A + (size_t)(m0 + crow) * K + k0 + cseg;
        async_cp16(ao, as);
        async_cp16(ao + 16, as + 8);
        unsigned bo = lds_addr(&Bbuf[buf][crow * 32 + cseg]);
        const bf16* wsrc = W + (size_t)(n0 + crow) * K + k0 + cseg;
        async_cp16(bo, wsrc);
        async_cp16(bo + 16, wsrc + 8);
    };

    v8f acc[2][4] = {};
    auto compute = [&](int cur) {
        v16bf a0 = load_a_frag(&Abuf[cur][(wr * 32) * 32], 32);
        v16bf a1 = load_a_frag(&Abuf[cur][(wr * 32 + 16) * 32], 32);
#pragma unroll
        for (int t = 0; t < 4; ++t) {
            v16bf b = load_b_frag(&Bbuf[cur][(wc * 64 + t * 16) * 32], 32);
            acc[0][t] = wmma_bf16(a0, b, acc[0][t]);
            acc[1][t] = wmma_bf16(a1, b, acc[1][t]);
        }
    };

    const int NS = K / 32;
    issue(0, 0);
    for (int k = 0; k < NS - 1; ++k) {
        issue((k + 1) & 1, (k + 1) * 32);
        WAIT_ASYNC(0x4);   // retire current buffer's 4 copies; next stays in flight
        __syncthreads();
        compute(k & 1);
        __syncthreads();
    }
    WAIT_ASYNC(0x0);
    __syncthreads();
    compute((NS - 1) & 1);

    // C layout: VGPR r -> row r (lanes 0-15) / r+8 (lanes 16-31); col = lane&15 + 16*t
    const int lane = tid & 31;
    const int col = lane & 15, hi = lane >> 4;
#pragma unroll
    for (int half = 0; half < 2; ++half)
#pragma unroll
        for (int t = 0; t < 4; ++t)
#pragma unroll
            for (int r = 0; r < 8; ++r)
                store(m0 + wr * 32 + half * 16 + r + 8 * hi,
                      n0 + wc * 64 + t * 16 + col, acc[half][t][r]);
}

__global__ void __launch_bounds__(256) gemm_qkv(const bf16* __restrict__ A,
                                                const bf16* __restrict__ W,
                                                bf16* __restrict__ Cmat,
                                                int K, int N) {
    gemm_body(A, W, K, [&](int m, int n, float v) {
        Cmat[(size_t)m * N + n] = (bf16)v;
    });
}

__global__ void __launch_bounds__(256) gemm_out(const bf16* __restrict__ A,
                                                const bf16* __restrict__ W,
                                                const float* __restrict__ x,
                                                const float* __restrict__ bias,
                                                float* __restrict__ out,
                                                int K, int N) {
    gemm_body(A, W, K, [&](int m, int n, float v) {
        size_t idx = (size_t)m * N + n;
        out[idx] = v + bias[n] + x[idx];
    });
}

// ---------------------------------------------------------------- flash attention
// grid.x = B*H (64), grid.y = S/64 (32); 128 threads = 4 waves, one 16-q tile each,
// sharing double-buffered async-staged K/V 32x64 tiles. Last iteration peeled.
__global__ void __launch_bounds__(128) attn_fwd(const bf16* __restrict__ qkv,
                                                bf16* __restrict__ out) {
    __shared__ __attribute__((aligned(16))) bf16 ldsK[2][32 * 64];
    __shared__ __attribute__((aligned(16))) bf16 ldsV[2][32 * 64];
    __shared__ __attribute__((aligned(16))) bf16 ldsP[4][16 * 32];

    const int tid = threadIdx.x, wave = tid >> 5, lane = tid & 31;
    const int b = blockIdx.x >> 4, h = blockIdx.x & 15;
    const int q0 = blockIdx.y * 64 + wave * 16;
    const int n = lane & 15, hi = lane >> 4;
    const size_t ld = 3 * C_;

    const bf16* Qb = qkv + ((size_t)(b * S_ + q0)) * ld + h * 192;
    const bf16* Kb = qkv + ((size_t)b * S_) * ld + h * 192 + 64;
    const bf16* Vb = qkv + ((size_t)b * S_) * ld + h * 192 + 128;

    v16bf aq0 = load_a_frag(Qb, ld);       // q channels 0..31
    v16bf aq1 = load_a_frag(Qb + 32, ld);  // q channels 32..63

    // staging: 32 rows x 64 ch per tile, 128 threads x 32B each
    const int srow = tid >> 2, sseg = (tid & 3) * 16;
    auto issueKV = [&](int buf, int kb) {
        const bf16* ks = Kb + (size_t)(kb * 32 + srow) * ld + sseg;
        unsigned ko = lds_addr(&ldsK[buf][srow * 64 + sseg]);
        async_cp16(ko, ks);
        async_cp16(ko + 16, ks + 8);
        const bf16* vs = Vb + (size_t)(kb * 32 + srow) * ld + sseg;
        unsigned vo = lds_addr(&ldsV[buf][srow * 64 + sseg]);
        async_cp16(vo, vs);
        async_cp16(vo + 16, vs + 8);
    };

    float mrow[8], lrow[8];
#pragma unroll
    for (int r = 0; r < 8; ++r) { mrow[r] = -1e30f; lrow[r] = 0.f; }
    v8f accO[4] = {};
    const float scale = 0.03125f;  // 1/sqrt(1024)

    auto step = [&](int cur) {
        // scores: 16 q x 32 k
        v8f s0 = {}, s1 = {};
        {
            v16bf bk;
            bk = load_b_frag(&ldsK[cur][0], 64);            s0 = wmma_bf16(aq0, bk, s0);
            bk = load_b_frag(&ldsK[cur][32], 64);           s0 = wmma_bf16(aq1, bk, s0);
            bk = load_b_frag(&ldsK[cur][16 * 64], 64);      s1 = wmma_bf16(aq0, bk, s1);
            bk = load_b_frag(&ldsK[cur][16 * 64 + 32], 64); s1 = wmma_bf16(aq1, bk, s1);
        }
        // online softmax; row = r + 8*hi spread across 16-lane group
#pragma unroll
        for (int r = 0; r < 8; ++r) {
            float a0 = s0[r] * scale, a1 = s1[r] * scale;
            float mx = fmaxf(a0, a1);
            mx = fmaxf(mx, __shfl_xor(mx, 1, 16));
            mx = fmaxf(mx, __shfl_xor(mx, 2, 16));
            mx = fmaxf(mx, __shfl_xor(mx, 4, 16));
            mx = fmaxf(mx, __shfl_xor(mx, 8, 16));
            float mnew = fmaxf(mrow[r], mx);
            float corr = __expf(mrow[r] - mnew);
            float p0 = __expf(a0 - mnew);
            float p1 = __expf(a1 - mnew);
            float rs = p0 + p1;
            rs += __shfl_xor(rs, 1, 16);
            rs += __shfl_xor(rs, 2, 16);
            rs += __shfl_xor(rs, 4, 16);
            rs += __shfl_xor(rs, 8, 16);
            lrow[r] = lrow[r] * corr + rs;
            mrow[r] = mnew;
#pragma unroll
            for (int t = 0; t < 4; ++t) accO[t][r] = accO[t][r] * corr;
            ldsP[wave][(r + 8 * hi) * 32 + n]      = (bf16)p0;
            ldsP[wave][(r + 8 * hi) * 32 + 16 + n] = (bf16)p1;
        }
        // P (C-frag layout in LDS) -> A-fragment; same-wave LDS ordering guarantees visibility
        v16bf ap;
        {
            const bf16* pp = &ldsP[wave][n * 32 + hi * 8];
            v8bf lo = *(const v8bf*)(pp);
            v8bf hq = *(const v8bf*)(pp + 16);
            ap = __builtin_shufflevector(lo, hq, 0,1,2,3,4,5,6,7,8,9,10,11,12,13,14,15);
        }
        // O += P @ V
#pragma unroll
        for (int t = 0; t < 4; ++t) {
            v16bf bv;
#pragma unroll
            for (int j = 0; j < 16; ++j)
                bv[j] = ldsV[cur][(hi * 8 + (j & 7) + (j >> 3) * 16) * 64 + t * 16 + n];
            accO[t] = wmma_bf16(ap, bv, accO[t]);
        }
    };

    const int NSK = S_ / 32;
    issueKV(0, 0);
    for (int kb = 0; kb < NSK - 1; ++kb) {
        issueKV((kb + 1) & 1, kb + 1);
        WAIT_ASYNC(0x4);
        __syncthreads();
        step(kb & 1);
        __syncthreads();
    }
    WAIT_ASYNC(0x0);
    __syncthreads();
    step((NSK - 1) & 1);

#pragma unroll
    for (int r = 0; r < 8; ++r) {
        float inv = 1.0f / lrow[r];
        int m = q0 + r + 8 * hi;
        size_t row = ((size_t)(b * S_ + m)) * C_ + h * CH_;
#pragma unroll
        for (int t = 0; t < 4; ++t)
            out[row + t * 16 + n] = (bf16)(accO[t][r] * inv);
    }
}

extern "C" void kernel_launch(void* const* d_in, const int* in_sizes, int n_in,
                              void* d_out, int out_size, void* d_ws, size_t ws_size,
                              hipStream_t stream) {
    const float* x     = (const float*)d_in[0];  // (4,2048,1024)
    const float* W_qkv = (const float*)d_in[1];  // (3072,1024)
    const float* W_out = (const float*)d_in[2];  // (1024,1024)
    const float* b_out = (const float*)d_in[3];  // (1024,)
    float* out = (float*)d_out;

    const int M = B_ * S_;    // 8192
    const int Nqkv = 3 * C_;  // 3072
    const int K = C_;         // 1024

    char* ws = (char*)d_ws;
    bf16* xb    = (bf16*)(ws);                             // 16 MB
    bf16* wqkvb = (bf16*)(ws + (size_t)16 * 1024 * 1024);  //  6 MB
    bf16* woutb = (bf16*)(ws + (size_t)22 * 1024 * 1024);  //  2 MB
    bf16* qkvb  = (bf16*)(ws + (size_t)24 * 1024 * 1024);  // 48 MB
    bf16* aob   = (bf16*)(ws + (size_t)72 * 1024 * 1024);  // 16 MB

    {
        int nx = M * C_;
        cvt_f32_bf16<<<(nx + 255) / 256, 256, 0, stream>>>(x, xb, nx);
        int nw = Nqkv * C_;
        cvt_f32_bf16<<<(nw + 255) / 256, 256, 0, stream>>>(W_qkv, wqkvb, nw);
        int no = C_ * C_;
        cvt_f32_bf16<<<(no + 255) / 256, 256, 0, stream>>>(W_out, woutb, no);
    }
    gemm_qkv<<<dim3(Nqkv / 128, M / 128), 256, 0, stream>>>(xb, wqkvb, qkvb, K, Nqkv);
    attn_fwd<<<dim3(B_ * H_, S_ / 64), 128, 0, stream>>>(qkvb, aob);
    gemm_out<<<dim3(C_ / 128, M / 128), 256, 0, stream>>>(aob, woutb, x, b_out, out, K, C_);
}
